// SkillModelVectorQuantized_326417514849
// MI455X (gfx1250) — matrix operand order repaired and checked by reading, not compiled, not run on hardware
//
#include <hip/hip_runtime.h>
#include <hip/hip_bf16.h>

// ---------------------------------------------------------------------------
// Types for CDNA5 WMMA (wave32): D(f32 16x16) = A(bf16 16x32) * B(bf16 32x16) + C
// ---------------------------------------------------------------------------
typedef __attribute__((ext_vector_type(16))) __bf16 v16bf;
typedef __attribute__((ext_vector_type(8)))  __bf16 v8bf;
typedef __attribute__((ext_vector_type(8)))  float  v8f;

#define TILE_ROWS 128
#define WAVES_PER_BLOCK 8
#define ST_X 136   // 128 K + 8 pad (bf16 elems) -> 272B rows, conflict-free b128 reads
#define ST_H 520   // 512 K + 8 pad             -> 1040B rows

enum { EPI_RELU_LDS = 0, EPI_LINEAR_GLOBAL = 1, EPI_SOFTPLUS_GLOBAL = 2 };

// ---------------------------------------------------------------------------
// One GEMM stage: [128 rows x Kdim] (bf16, LDS) x [Kdim x Nout] (bf16, global,
// pre-swizzled into WMMA B-fragments) -> epilogue.
// Wave tiling: each task = 64 rows x 16 cols (4 A-frags reuse 1 B-frag).
// Unroll capped at 2 to stay under 256 VGPRs (no spills, no VGPR-MSB mode).
// ---------------------------------------------------------------------------
static __device__ __forceinline__ void gemm_stage(
    const __bf16* __restrict__ lds_in, int stride_in, int Kdim,
    const __bf16* __restrict__ Wp, const float* __restrict__ bias,
    int Nout, int n_real, int mode,
    __bf16* __restrict__ lds_out, int stride_out,
    float* __restrict__ gout, int ld_out, long row_base)
{
    const int wave  = threadIdx.x >> 5;
    const int lane  = threadIdx.x & 31;
    const int lhalf = lane >> 4;     // 0 or 1
    const int lmod  = lane & 15;
    const int KS    = Kdim >> 5;     // k-slabs of 32
    const int ntasks = (Nout >> 4) * 2;

    for (int t = wave; t < ntasks; t += WAVES_PER_BLOCK) {
        const int ct = t >> 1;          // 16-col tile
        const int r0 = (t & 1) * 64;    // row half of the 128-row tile

        v8f acc[4] = {};

        // B fragments: lane-contiguous 32B blocks, streamed from global (L2-hot)
        const __bf16* bp  = Wp + (((long)ct * KS) * 32 + lane) * 16;
        // A fragments: lane (0-15): K k..k+15 ; lane (16-31): +16
        const __bf16* ap0 = lds_in + (r0 + lmod) * stride_in + lhalf * 16;

#pragma unroll 2
        for (int ks = 0; ks < KS; ++ks) {
            v16bf bfrag = *(const v16bf*)bp;        // 32B global load
            // pull the B fragment two k-slabs ahead toward L0/L2 (global_prefetch_b8)
            __builtin_prefetch(bp + 2 * 32 * 16, 0, 3);
            bp += 32 * 16;
            const __bf16* ap = ap0 + ks * 32;
#pragma unroll
            for (int i = 0; i < 4; ++i) {
                v8bf lo = *(const v8bf*)(ap + i * 16 * stride_in);        // ds b128
                v8bf hi = *(const v8bf*)(ap + i * 16 * stride_in + 8);    // ds b128
                v16bf afrag = __builtin_shufflevector(
                    lo, hi, 0,1,2,3,4,5,6,7,8,9,10,11,12,13,14,15);
                acc[i] = __builtin_amdgcn_wmma_f32_16x16x32_bf16(
                    false, afrag, false, bfrag, (short)0, acc[i], false, false);
            }
        }

        // Epilogue. C/D layout: VGPR j, lane<16 -> M=j,N=lane ; lane>=16 -> M=j+8
        const int ncol = ct * 16 + lmod;
        const float bv = (ncol < n_real) ? bias[ncol] : 0.f;
#pragma unroll
        for (int i = 0; i < 4; ++i) {
#pragma unroll
            for (int j = 0; j < 8; ++j) {
                const int m = r0 + i * 16 + (lhalf ? j + 8 : j);
                float v = acc[i][j] + bv;
                if (mode == EPI_RELU_LDS) {
                    v = fmaxf(v, 0.f);
                    lds_out[m * stride_out + ncol] = (__bf16)v;
                } else if (ncol < n_real) {
                    if (mode == EPI_SOFTPLUS_GLOBAL)
                        v = fmaxf(v, 0.f) + log1pf(expf(-fabsf(v)));   // stable softplus
                    gout[(row_base + m) * (long)ld_out + ncol] = v;
                }
            }
        }
    }
}

// ---------------------------------------------------------------------------
// Fused MLP: x = concat(state_row, zq) -> trunk (2 layers) -> 2 heads.
// Serves both the policy path (131072 rows) and decoder path (1024 rows).
// ---------------------------------------------------------------------------
struct MlpArgs {
    const float* states; long srow_stride; int bshift;
    const float* zq;
    const __bf16 *w1, *w2, *wm1, *wm2, *ws1, *ws2;
    const float  *b1, *b2, *bm1, *bm2, *bs1, *bs2;
    float* out_mean; float* out_sig;
    int ld_out; int n_real_head; int nout_head;
};

__global__ void __launch_bounds__(256) fused_mlp_kernel(MlpArgs A) {
    extern __shared__ char smem[];
    __bf16* xs = (__bf16*)smem;                 // 128 x 136  (34816 B)
    __bf16* bA = xs + TILE_ROWS * ST_X;         // 128 x 520  (133120 B)
    __bf16* bB = bA + TILE_ROWS * ST_H;         // 128 x 520  (133120 B)
    const long row_base = (long)blockIdx.x * TILE_ROWS;

    // Stage 0: build bf16 input tile [state(60) | zq(64) | zero-pad(to 128+)]
    for (int e = threadIdx.x; e < TILE_ROWS * ST_X; e += blockDim.x) {
        const int r = e / ST_X, c = e % ST_X;
        const long gr = row_base + r;
        float v = 0.f;
        if (c < 60)       v = A.states[gr * A.srow_stride + c];
        else if (c < 124) v = A.zq[(gr >> A.bshift) * 64 + (c - 60)];
        xs[e] = (__bf16)v;
    }
    __syncthreads();

    gemm_stage(xs, ST_X, 128, A.w1, A.b1, 512, 512, EPI_RELU_LDS, bA, ST_H, nullptr, 0, 0);
    __syncthreads();
    gemm_stage(bA, ST_H, 512, A.w2, A.b2, 512, 512, EPI_RELU_LDS, bB, ST_H, nullptr, 0, 0);
    __syncthreads();
    gemm_stage(bB, ST_H, 512, A.wm1, A.bm1, 512, 512, EPI_RELU_LDS, bA, ST_H, nullptr, 0, 0);
    __syncthreads();
    gemm_stage(bA, ST_H, 512, A.wm2, A.bm2, A.nout_head, A.n_real_head,
               EPI_LINEAR_GLOBAL, nullptr, 0, A.out_mean, A.ld_out, row_base);
    __syncthreads();
    gemm_stage(bB, ST_H, 512, A.ws1, A.bs1, 512, 512, EPI_RELU_LDS, bA, ST_H, nullptr, 0, 0);
    __syncthreads();
    gemm_stage(bA, ST_H, 512, A.ws2, A.bs2, A.nout_head, A.n_real_head,
               EPI_SOFTPLUS_GLOBAL, nullptr, 0, A.out_sig, A.ld_out, row_base);
}

// ---------------------------------------------------------------------------
// Weight prep: f32 [Kreal x Nreal] row-major -> bf16 WMMA B-fragment layout,
// zero-padded to [Kpad x Npad]. Fragment: [nt][ks][lane][16 bf16].
//   lane = (n%16) + 16*(kk>=16);  slot bf16 index = kk%16
// ---------------------------------------------------------------------------
__global__ void prep_w_kernel(const float* __restrict__ src, __bf16* __restrict__ dst,
                              int Kreal, int Nreal, int Kpad, int Npad) {
    const int total = Kpad * Npad;
    for (int e = blockIdx.x * blockDim.x + threadIdx.x; e < total;
         e += gridDim.x * blockDim.x) {
        const int n = e % Npad, k = e / Npad;
        const float v = (k < Kreal && n < Nreal) ? src[k * Nreal + n] : 0.f;
        const int nt = n >> 4, nl = n & 15, ks = k >> 5, kk = k & 31;
        const int lane = nl + ((kk >> 4) << 4);
        const long slot = (((long)nt * (Kpad >> 5) + ks) * 32 + lane) * 16 + (kk & 15);
        dst[slot] = (__bf16)v;
    }
}

// ---------------------------------------------------------------------------
// VQ: one wave per batch row. Each lane scans 32 codes serially, then a
// cross-lane min+argmin reduction; writes zq row and accumulates 1.25*MSE.
// ---------------------------------------------------------------------------
__global__ void __launch_bounds__(256) vq_kernel(
    const float* __restrict__ z_e, const float* __restrict__ codebook,
    float* __restrict__ zq, float* __restrict__ loss_out)
{
    const int wave = threadIdx.x >> 5;
    const int lane = threadIdx.x & 31;
    const int b = blockIdx.x * WAVES_PER_BLOCK + wave;   // 128 blocks * 8 = 1024 rows

    float ze[64];
#pragma unroll
    for (int d = 0; d < 64; ++d) ze[d] = z_e[(long)b * 64 + d];

    float bestd = 1e30f; int bestk = 0;
    for (int k = lane; k < 1024; k += 32) {
        const float* c = codebook + (long)k * 64;
        float d = 0.f;
#pragma unroll
        for (int dd = 0; dd < 64; ++dd) { float t = ze[dd] - c[dd]; d = fmaf(t, t, d); }
        if (d < bestd) { bestd = d; bestk = k; }
    }
#pragma unroll
    for (int off = 16; off >= 1; off >>= 1) {
        float od = __shfl_xor(bestd, off, 32);
        int   ok = __shfl_xor(bestk, off, 32);
        if (od < bestd || (od == bestd && ok < bestk)) { bestd = od; bestk = ok; }
    }
    // all lanes agree on bestk: emit zq row + loss partial (2 dims/lane)
    const float* c = codebook + (long)bestk * 64;
    float part = 0.f;
    for (int d = lane; d < 64; d += 32) {
        const float cv = c[d];
        zq[(long)b * 64 + d] = cv;
        const float t = ze[d] - cv;
        part = fmaf(t, t, part);
    }
#pragma unroll
    for (int off = 16; off >= 1; off >>= 1) part += __shfl_xor(part, off, 32);
    if (lane == 0) atomicAdd(loss_out, part * (1.25f / (1024.0f * 64.0f)));
}

__global__ void zero_loss_kernel(float* p) { if (threadIdx.x == 0) *p = 0.f; }

// ---------------------------------------------------------------------------
// Host launcher
// ---------------------------------------------------------------------------
extern "C" void kernel_launch(void* const* d_in, const int* in_sizes, int n_in,
                              void* d_out, int out_size, void* d_ws, size_t ws_size,
                              hipStream_t stream) {
    (void)in_sizes; (void)n_in; (void)out_size; (void)ws_size;
    const float* states   = (const float*)d_in[0];
    const float* z_e      = (const float*)d_in[1];
    const float* codebook = (const float*)d_in[2];
    const float* dw1 = (const float*)d_in[3];  const float* db1 = (const float*)d_in[4];
    const float* dw2 = (const float*)d_in[5];  const float* db2 = (const float*)d_in[6];
    const float* dwm1= (const float*)d_in[7];  const float* dbm1= (const float*)d_in[8];
    const float* dwm2= (const float*)d_in[9];  const float* dbm2= (const float*)d_in[10];
    const float* dws1= (const float*)d_in[11]; const float* dbs1= (const float*)d_in[12];
    const float* dws2= (const float*)d_in[13]; const float* dbs2= (const float*)d_in[14];
    const float* pw1 = (const float*)d_in[15]; const float* pb1 = (const float*)d_in[16];
    const float* pw2 = (const float*)d_in[17]; const float* pb2 = (const float*)d_in[18];
    const float* pwm1= (const float*)d_in[19]; const float* pbm1= (const float*)d_in[20];
    const float* pwm2= (const float*)d_in[21]; const float* pbm2= (const float*)d_in[22];
    const float* pws1= (const float*)d_in[23]; const float* pbs1= (const float*)d_in[24];
    const float* pws2= (const float*)d_in[25]; const float* pbs2= (const float*)d_in[26];

    float* out = (float*)d_out;                  // (sT_mean, sT_sig, a_mean, a_sig, loss)
    float* sT_mean = out;                        // 1024*60
    float* sT_sig  = out + 1024 * 60;            // 1024*60
    float* a_mean  = out + 2 * 1024 * 60;        // 1024*128*8
    float* a_sig   = a_mean + 1024 * 128 * 8;    // 1024*128*8
    float* loss    = a_sig  + 1024 * 128 * 8;    // scalar

    // --- workspace carving ---
    char* ws = (char*)d_ws; size_t off = 0;
    auto carve = [&](size_t bytes) -> void* {
        void* p = ws + off; off += (bytes + 255) & ~(size_t)255; return p;
    };
    float* zq = (float*)carve(1024 * 64 * sizeof(float));

    auto prep = [&](const float* src, int Kr, int Nr, int Kp, int Np) -> __bf16* {
        __bf16* dst = (__bf16*)carve((size_t)Kp * Np * 2);
        int total = Kp * Np;
        int grid = (total + 255) / 256; if (grid > 1024) grid = 1024;
        prep_w_kernel<<<grid, 256, 0, stream>>>(src, dst, Kr, Nr, Kp, Np);
        return dst;
    };

    // VQ first (policy & decoder both consume zq)
    zero_loss_kernel<<<1, 32, 0, stream>>>(loss);
    vq_kernel<<<128, 256, 0, stream>>>(z_e, codebook, zq, loss);

    // Weight conversions (bf16, WMMA B-fragment swizzle, zero padding)
    __bf16* dw1p  = prep(dw1 , 124, 512, 128, 512);
    __bf16* dw2p  = prep(dw2 , 512, 512, 512, 512);
    __bf16* dwm1p = prep(dwm1, 512, 512, 512, 512);
    __bf16* dwm2p = prep(dwm2, 512,  60, 512,  64);
    __bf16* dws1p = prep(dws1, 512, 512, 512, 512);
    __bf16* dws2p = prep(dws2, 512,  60, 512,  64);
    __bf16* pw1p  = prep(pw1 , 124, 512, 128, 512);
    __bf16* pw2p  = prep(pw2 , 512, 512, 512, 512);
    __bf16* pwm1p = prep(pwm1, 512, 512, 512, 512);
    __bf16* pwm2p = prep(pwm2, 512,   8, 512,  16);
    __bf16* pws1p = prep(pws1, 512, 512, 512, 512);
    __bf16* pws2p = prep(pws2, 512,   8, 512,  16);

    const size_t smem = (size_t)(TILE_ROWS * ST_X + 2 * TILE_ROWS * ST_H) * 2; // 301056 B

    // Policy path: row gr = b*128 + t ; states stride 60 ; zq index = gr>>7
    MlpArgs P;
    P.states = states; P.srow_stride = 60;  P.bshift = 7; P.zq = zq;
    P.w1 = pw1p; P.w2 = pw2p; P.wm1 = pwm1p; P.wm2 = pwm2p; P.ws1 = pws1p; P.ws2 = pws2p;
    P.b1 = pb1;  P.b2 = pb2;  P.bm1 = pbm1;  P.bm2 = pbm2;  P.bs1 = pbs1;  P.bs2 = pbs2;
    P.out_mean = a_mean; P.out_sig = a_sig; P.ld_out = 8; P.n_real_head = 8; P.nout_head = 16;
    fused_mlp_kernel<<<1024, 256, smem, stream>>>(P);

    // Decoder path: row gr = b ; states stride T*S = 7680 ; zq index = gr
    MlpArgs D;
    D.states = states; D.srow_stride = 7680; D.bshift = 0; D.zq = zq;
    D.w1 = dw1p; D.w2 = dw2p; D.wm1 = dwm1p; D.wm2 = dwm2p; D.ws1 = dws1p; D.ws2 = dws2p;
    D.b1 = db1;  D.b2 = db2;  D.bm1 = dbm1;  D.bm2 = dbm2;  D.bs1 = dbs1;  D.bs2 = dbs2;
    D.out_mean = sT_mean; D.out_sig = sT_sig; D.ld_out = 60; D.n_real_head = 60; D.nout_head = 64;
    fused_mlp_kernel<<<8, 256, smem, stream>>>(D);
}
